// PatchGraphAttention_43301860278571
// MI455X (gfx1250) — matrix operand order repaired and checked by reading, not compiled, not run on hardware
//
#include <hip/hip_runtime.h>

typedef __attribute__((ext_vector_type(16))) _Float16 v16h;
typedef __attribute__((ext_vector_type(8)))  _Float16 v8h;
typedef __attribute__((ext_vector_type(4)))  _Float16 v4h;
typedef __attribute__((ext_vector_type(8)))  float    v8f;

#define D_MODEL 1024
#define NHEAD   16
#define D_HEAD  64
#define BATCH   4
#define SEQ     2048
#define ROWS    (BATCH * SEQ)   // 8192

union F16Frag { v16h v; v8h h[2]; };

__device__ __forceinline__ v8f wmma16(v16h a, v16h b, v8f c) {
  // D = A(16x32) * B(32x16) + C, f32 accumulate
  return __builtin_amdgcn_wmma_f32_16x16x32_f16(false, a, false, b, (short)0, c,
                                                false, false);
}

// A-matrix fragment (16x32 f16): lane l -> row (l&15), K = base..base+7 and
// base+16..base+23, base = (l>>4)*8.  Source row-major, leading dim ld.
__device__ __forceinline__ v16h load_fragA(const _Float16* __restrict__ p, int ld,
                                           int row0, int kbase, int lane) {
  int r  = row0 + (lane & 15);
  int kb = kbase + ((lane >> 4) << 3);
  const _Float16* q = p + (size_t)r * ld + kb;
  F16Frag f;
  f.h[0] = *(const v8h*)(q);
  f.h[1] = *(const v8h*)(q + 16);
  return f.v;
}

// B-matrix fragment (32x16 f16): lane l -> column (l&15) of B (= row of B^T),
// K = base..base+7 and base+16..base+23, base = kbase + (l>>4)*8.
__device__ __forceinline__ v16h load_fragB(const _Float16* __restrict__ p, int ld,
                                           int col0, int kbase, int lane) {
  int n  = col0 + (lane & 15);
  int kb = kbase + ((lane >> 4) << 3);
  const _Float16* q = p + (size_t)n * ld + kb;
  F16Frag f;
  f.h[0] = *(const v8h*)(q);
  f.h[1] = *(const v8h*)(q + 16);
  return f.v;
}

// ---------------------------------------------------------------- fp32 -> f16
__global__ __launch_bounds__(256) void cvt_f2h_kernel(const float* __restrict__ in,
                                                      _Float16* __restrict__ out,
                                                      int n4) {
  int i = blockIdx.x * blockDim.x + threadIdx.x;
  if (i < n4) {
    float4 v = ((const float4*)in)[i];
    v4h o;
    o[0] = (_Float16)v.x; o[1] = (_Float16)v.y;
    o[2] = (_Float16)v.z; o[3] = (_Float16)v.w;
    ((v4h*)out)[i] = o;
  }
}

// ------------------------------------------------- QKV projection: C = A*W^T+b
// 32x64 output per wave: 2 A-frags + 4 B-frags -> 8 WMMAs per k-step.
// vmode==0: store per-head row-major  out[bh][m][64]
// vmode==1: store per-head transposed out[bh][64][m]   (for V)
__global__ __launch_bounds__(128) void gemm_qkv_kernel(
    const _Float16* __restrict__ A, const _Float16* __restrict__ W,
    const float* __restrict__ bias, _Float16* __restrict__ out, int vmode) {
  const int lane = threadIdx.x & 31;
  const int wave = threadIdx.x >> 5;
  const int n0 = blockIdx.x * 64;
  const int m0 = (blockIdx.y * 4 + wave) * 32;

  v8f acc[2][4] = {};
  for (int k = 0; k < D_MODEL; k += 32) {
    v16h a0 = load_fragA(A, D_MODEL, m0,      k, lane);
    v16h a1 = load_fragA(A, D_MODEL, m0 + 16, k, lane);
#pragma unroll
    for (int j = 0; j < 4; ++j) {
      v16h b = load_fragB(W, D_MODEL, n0 + j * 16, k, lane);
      acc[0][j] = wmma16(a0, b, acc[0][j]);
      acc[1][j] = wmma16(a1, b, acc[1][j]);
    }
  }
  const int lid  = lane & 15;
  const int hup8 = (lane >> 4) << 3;
#pragma unroll
  for (int t = 0; t < 2; ++t) {
#pragma unroll
    for (int j = 0; j < 4; ++j) {
      int n = n0 + j * 16 + lid;
      float bn = bias[n];
      int h = n >> 6, d = n & 63;
#pragma unroll
      for (int i = 0; i < 8; ++i) {
        int mg = m0 + t * 16 + i + hup8;
        int bb = mg >> 11, m = mg & (SEQ - 1);
        float val = acc[t][j][i] + bn;
        size_t idx;
        if (vmode)
          idx = ((size_t)(bb * NHEAD + h) * D_HEAD + d) * SEQ + m;
        else
          idx = ((size_t)(bb * NHEAD + h) * SEQ + m) * D_HEAD + d;
        out[idx] = (_Float16)val;
      }
    }
  }
}

// ------------------------------------------------------- fused attention core
// One wave handles one (bh, 32-query strip = 2 tiles); flash-style online
// softmax over 32-key chunks.  K/V fragments are shared across the two query
// tiles: 16 WMMAs per 8 fragment loads per chunk.
__global__ __launch_bounds__(32) void attn_kernel(
    const _Float16* __restrict__ Q, const _Float16* __restrict__ K,
    const _Float16* __restrict__ Vt, const int* __restrict__ pmask,
    _Float16* __restrict__ AO) {
  __shared__ __align__(16) _Float16 ldsP[32 * 32];   // 2 KB: P for both q-tiles
  const int lane = threadIdx.x;
  const int qt = blockIdx.x, bh = blockIdx.y;
  const int b = bh >> 4, h = bh & 15;
  const _Float16* Qh = Q  + (size_t)bh * SEQ * D_HEAD;
  const _Float16* Kh = K  + (size_t)bh * SEQ * D_HEAD;
  const _Float16* Vh = Vt + (size_t)bh * D_HEAD * SEQ;
  const int* pm = pmask + b * SEQ;
  const int q0 = qt * 32;
  const int lid = lane & 15, hup = lane >> 4;

  v16h qa[2][2];
#pragma unroll
  for (int t = 0; t < 2; ++t) {
    qa[t][0] = load_fragA(Qh, D_HEAD, q0 + t * 16, 0, lane);
    qa[t][1] = load_fragA(Qh, D_HEAD, q0 + t * 16, 32, lane);
  }

  v8f o[2][4] = {};
  float mI[2][8], lI[2][8];
#pragma unroll
  for (int t = 0; t < 2; ++t)
#pragma unroll
    for (int i = 0; i < 8; ++i) { mI[t][i] = -__builtin_inff(); lI[t][i] = 0.f; }

  for (int kc = 0; kc < SEQ; kc += 32) {
    // ---- scores: 2 query tiles x 2 key tiles, k-dim 64 = 2 WMMAs each
    v16h kb00 = load_fragB(Kh, D_HEAD, kc,      0,  lane);
    v16h kb01 = load_fragB(Kh, D_HEAD, kc,      32, lane);
    v16h kb10 = load_fragB(Kh, D_HEAD, kc + 16, 0,  lane);
    v16h kb11 = load_fragB(Kh, D_HEAD, kc + 16, 32, lane);

    const int k0 = kc + lid, k1 = kc + 16 + lid;
    const bool msk0 = (pm[k0] == 0), msk1 = (pm[k1] == 0);

#pragma unroll
    for (int t = 0; t < 2; ++t) {
      v8f s0 = {}, s1 = {};
      s0 = wmma16(qa[t][0], kb00, s0);
      s0 = wmma16(qa[t][1], kb01, s0);
      s1 = wmma16(qa[t][0], kb10, s1);
      s1 = wmma16(qa[t][1], kb11, s1);

      float p0[8], p1[8];
#pragma unroll
      for (int i = 0; i < 8; ++i) {
        int qg = q0 + t * 16 + i + hup * 8;
        // scale 1/sqrt(64) + analytic time bias (0 on diag, ln(1e-12) off),
        // then key-mask replacement with -1e9 (matches reference order).
        float a0 = msk0 ? -1e9f : s0[i] * 0.125f + (qg == k0 ? 0.f : -27.6310211f);
        float a1 = msk1 ? -1e9f : s1[i] * 0.125f + (qg == k1 ? 0.f : -27.6310211f);
        float mx = fmaxf(a0, a1);
        mx = fmaxf(mx, __shfl_xor(mx, 1));
        mx = fmaxf(mx, __shfl_xor(mx, 2));
        mx = fmaxf(mx, __shfl_xor(mx, 4));
        mx = fmaxf(mx, __shfl_xor(mx, 8));
        float mn   = fmaxf(mI[t][i], mx);
        float corr = __expf(mI[t][i] - mn);
        mI[t][i] = mn;
        float e0 = __expf(a0 - mn), e1 = __expf(a1 - mn);
        p0[i] = e0; p1[i] = e1;
        float rs = e0 + e1;
        rs += __shfl_xor(rs, 1);
        rs += __shfl_xor(rs, 2);
        rs += __shfl_xor(rs, 4);
        rs += __shfl_xor(rs, 8);
        lI[t][i] = lI[t][i] * corr + rs;
#pragma unroll
        for (int j = 0; j < 4; ++j) o[t][j][i] *= corr;
      }
      // C-layout -> A-layout repack of P through LDS
#pragma unroll
      for (int i = 0; i < 8; ++i) {
        int m = t * 16 + i + hup * 8;
        ldsP[m * 32 + lid]      = (_Float16)p0[i];
        ldsP[m * 32 + 16 + lid] = (_Float16)p1[i];
      }
    }
    __syncthreads();
    F16Frag pf[2];
#pragma unroll
    for (int t = 0; t < 2; ++t) {
      pf[t].h[0] = *(const v8h*)&ldsP[(t * 16 + lid) * 32 + hup * 8];
      pf[t].h[1] = *(const v8h*)&ldsP[(t * 16 + lid) * 32 + hup * 8 + 16];
    }
    __syncthreads();
    // ---- O += P * V : V fragments shared by both query tiles
#pragma unroll
    for (int j = 0; j < 4; ++j) {
      v16h vb = load_fragB(Vh, SEQ, j * 16, kc, lane);
      o[0][j] = wmma16(pf[0].v, vb, o[0][j]);
      o[1][j] = wmma16(pf[1].v, vb, o[1][j]);
    }
  }

#pragma unroll
  for (int t = 0; t < 2; ++t)
#pragma unroll
    for (int i = 0; i < 8; ++i) {
      float inv = 1.f / lI[t][i];
      int mg = q0 + t * 16 + i + hup * 8;
      size_t base = ((size_t)(b * SEQ + mg)) * D_MODEL + h * D_HEAD;
#pragma unroll
      for (int j = 0; j < 4; ++j)
        AO[base + j * 16 + lid] = (_Float16)(o[t][j][i] * inv);
    }
}

// --------------------------------------- output projection + bias + patch mask
__global__ __launch_bounds__(128) void gemm_out_kernel(
    const _Float16* __restrict__ A, const _Float16* __restrict__ W,
    const float* __restrict__ bias, const int* __restrict__ pmask,
    float* __restrict__ out) {
  const int lane = threadIdx.x & 31;
  const int wave = threadIdx.x >> 5;
  const int n0 = blockIdx.x * 64;
  const int m0 = (blockIdx.y * 4 + wave) * 32;

  v8f acc[2][4] = {};
  for (int k = 0; k < D_MODEL; k += 32) {
    v16h a0 = load_fragA(A, D_MODEL, m0,      k, lane);
    v16h a1 = load_fragA(A, D_MODEL, m0 + 16, k, lane);
#pragma unroll
    for (int j = 0; j < 4; ++j) {
      v16h b = load_fragB(W, D_MODEL, n0 + j * 16, k, lane);
      acc[0][j] = wmma16(a0, b, acc[0][j]);
      acc[1][j] = wmma16(a1, b, acc[1][j]);
    }
  }
  const int lid  = lane & 15;
  const int hup8 = (lane >> 4) << 3;
#pragma unroll
  for (int t = 0; t < 2; ++t) {
#pragma unroll
    for (int j = 0; j < 4; ++j) {
      int n = n0 + j * 16 + lid;
      float bn = bias[n];
#pragma unroll
      for (int i = 0; i < 8; ++i) {
        int mg = m0 + t * 16 + i + hup8;
        float val = acc[t][j][i] + bn;
        out[(size_t)mg * D_MODEL + n] = val * (float)pmask[mg];
      }
    }
  }
}

extern "C" void kernel_launch(void* const* d_in, const int* in_sizes, int n_in,
                              void* d_out, int out_size, void* d_ws, size_t ws_size,
                              hipStream_t stream) {
  const float* H   = (const float*)d_in[0];
  const int* pmask = (const int*)d_in[1];
  const float* Wq  = (const float*)d_in[2];
  const float* bq  = (const float*)d_in[3];
  const float* Wk  = (const float*)d_in[4];
  const float* bk  = (const float*)d_in[5];
  const float* Wv  = (const float*)d_in[6];
  const float* bv  = (const float*)d_in[7];
  const float* Wo  = (const float*)d_in[8];
  const float* bo  = (const float*)d_in[9];
  float* out = (float*)d_out;

  // workspace carve-up (~88 MB)
  char* p = (char*)d_ws;
  _Float16* H16  = (_Float16*)p; p += (size_t)ROWS * D_MODEL * 2;
  _Float16* Wq16 = (_Float16*)p; p += (size_t)D_MODEL * D_MODEL * 2;
  _Float16* Wk16 = (_Float16*)p; p += (size_t)D_MODEL * D_MODEL * 2;
  _Float16* Wv16 = (_Float16*)p; p += (size_t)D_MODEL * D_MODEL * 2;
  _Float16* Wo16 = (_Float16*)p; p += (size_t)D_MODEL * D_MODEL * 2;
  _Float16* Q16  = (_Float16*)p; p += (size_t)ROWS * D_MODEL * 2;
  _Float16* K16  = (_Float16*)p; p += (size_t)ROWS * D_MODEL * 2;
  _Float16* Vt16 = (_Float16*)p; p += (size_t)ROWS * D_MODEL * 2;
  _Float16* AO16 = (_Float16*)p; p += (size_t)ROWS * D_MODEL * 2;

  const int nH4 = ROWS * D_MODEL / 4;     // 2M float4
  const int nW4 = D_MODEL * D_MODEL / 4;  // 256K float4
  cvt_f2h_kernel<<<nH4 / 256, 256, 0, stream>>>(H,  H16,  nH4);
  cvt_f2h_kernel<<<nW4 / 256, 256, 0, stream>>>(Wq, Wq16, nW4);
  cvt_f2h_kernel<<<nW4 / 256, 256, 0, stream>>>(Wk, Wk16, nW4);
  cvt_f2h_kernel<<<nW4 / 256, 256, 0, stream>>>(Wv, Wv16, nW4);
  cvt_f2h_kernel<<<nW4 / 256, 256, 0, stream>>>(Wo, Wo16, nW4);

  dim3 gg(D_MODEL / 64, ROWS / 128);  // (16, 64), 4 waves/block, 32 rows/wave
  gemm_qkv_kernel<<<gg, 128, 0, stream>>>(H16, Wq16, bq, Q16, 0);
  gemm_qkv_kernel<<<gg, 128, 0, stream>>>(H16, Wk16, bk, K16, 0);
  gemm_qkv_kernel<<<gg, 128, 0, stream>>>(H16, Wv16, bv, Vt16, 1);

  attn_kernel<<<dim3(SEQ / 32, BATCH * NHEAD), 32, 0, stream>>>(
      Q16, K16, Vt16, pmask, AO16);

  gemm_out_kernel<<<gg, 128, 0, stream>>>(AO16, Wo16, bo, pmask, out);
}
